// PerforatedConv2d_21612275433669
// MI455X (gfx1250) — compile-verified, matrix operand order checked by reading, not run on hardware
//
#include <hip/hip_runtime.h>
#include <hip/hip_bf16.h>

typedef _Float16 v8h  __attribute__((ext_vector_type(8)));
typedef _Float16 v16h __attribute__((ext_vector_type(16)));
typedef float    v8f  __attribute__((ext_vector_type(8)));
typedef unsigned int u32x4 __attribute__((ext_vector_type(4)));
typedef int          i32x8 __attribute__((ext_vector_type(8)));
typedef int          i32x4 __attribute__((ext_vector_type(4)));

#define NIMG 32
#define CIN  128
#define COUT 256
#define HW   112
#define OH   56
#define OW   56
#define KTOT 1152          // 9 * 128
#define BM   128
#define BN   128
#define BK   32
#define LDS_STRIDE 40      // f16 elements per row (80 B; 64 B data + 16 B pad -> conflict-free b128)

#if defined(__has_builtin)
#if __has_builtin(__builtin_amdgcn_tensor_load_to_lds) && __has_builtin(__builtin_amdgcn_s_wait_tensorcnt)
#define USE_TDM 1
#endif
#endif
#ifndef USE_TDM
#define USE_TDM 0
#endif

// ---------------------------------------------------------------------------
// Prepass: convert weights f32 -> f16 and reorder K as (ky*3+kx)*128 + ci so
// each 32-wide K-chunk has a single (ky,kx) and contiguous ci.
// wf[co][k] with k = khw*128 + ci ;  source w[co][ci][ky][kx]
// ---------------------------------------------------------------------------
__global__ __launch_bounds__(256)
void repack_weight_f16(const float* __restrict__ w, _Float16* __restrict__ wf) {
    int idx = blockIdx.x * 256 + threadIdx.x;
    if (idx >= COUT * KTOT) return;
    int co  = idx / KTOT;
    int k   = idx - co * KTOT;
    int khw = k >> 7;          // 0..8 == ky*3+kx
    int ci  = k & 127;
    wf[idx] = (_Float16)w[co * KTOT + ci * 9 + khw];
}

// ---------------------------------------------------------------------------
// Implicit-GEMM perforated conv with f16 WMMA, f32 accumulate.
// Grid: x = spatial tiles (NIMG*OH*OW / BN = 784), y = Cout tiles (2).
// A tile (weights, pure f16 global->LDS copy) staged by the Tensor Data Mover;
// B tile (input patches) gathered + converted by the waves.
// ---------------------------------------------------------------------------
__global__ __launch_bounds__(256)
void perf_conv_wmma(const float* __restrict__ x,
                    const _Float16* __restrict__ wf,
                    const float* __restrict__ bias,
                    float* __restrict__ out) {
    __shared__ _Float16 ldsA[BM * LDS_STRIDE];   // [co_local][k_chunk]  10.0 KB
    __shared__ _Float16 ldsB[BN * LDS_STRIDE];   // [s_local ][k_chunk]  10.0 KB

    const int tid    = threadIdx.x;
    const int wave   = tid >> 5;
    const int lane   = tid & 31;
    const int ln     = lane & 15;
    const int hi16   = lane >> 4;          // 0: lanes 0-15, 1: lanes 16-31
    const int wave_m = wave >> 1;          // 0..3 -> rows wave_m*32 + {0,16}
    const int wave_n = wave & 1;           // 0..1 -> cols wave_n*64 + {0,16,32,48}

    const int s_blk  = blockIdx.x * BN;
    const int co_blk = blockIdx.y * BM;

    // staging thread mapping: 256 threads cover 128 rows x 2 halves of 32-wide K row
    const int st_row  = tid & 127;
    const int st_half = tid >> 7;

    // per-thread small-grid coordinates for B staging
    const int sb   = s_blk + st_row;
    const int simg = sb / (OH * OW);
    int srem       = sb - simg * (OH * OW);
    const int soy  = srem / OW;
    const int sox  = srem - soy * OW;

#if USE_TDM
    const unsigned int ldsA_off = (unsigned int)(uintptr_t)(&ldsA[0]);  // LDS byte offset
    const unsigned long long wf_base =
        (unsigned long long)(uintptr_t)(wf + (size_t)co_blk * KTOT);
#endif

    v8f acc[2][4];
    #pragma unroll
    for (int i = 0; i < 2; ++i)
        #pragma unroll
        for (int j = 0; j < 4; ++j)
            acc[i][j] = v8f{};

    for (int kc = 0; kc < KTOT / BK; ++kc) {
        const int khw = kc >> 2;                 // (ky*3+kx); 4 chunks per tap
        const int ci0 = (kc & 3) * 32;
        const int ky  = khw / 3;
        const int kx  = khw - 3 * ky;

        __syncthreads();                         // LDS reuse fence

        // ---- stage A: 128 x 32 f16 weight tile ----
#if USE_TDM
        if (wave == 0) {
            // D# group0: count=1 | lds_addr | global_addr[56:0] | type=2
            const unsigned long long ga = wf_base + (unsigned long long)kc * BK * 2ull;
            u32x4 g0;
            g0.x = 1u;                                   // count=1, user descriptor
            g0.y = ldsA_off;                             // LDS destination (bytes)
            g0.z = (unsigned int)ga;                     // global_addr[31:0]
            g0.w = (unsigned int)((ga >> 32) & 0x1FFFFFFull) | (2u << 30); // [56:32] | type=2
            // D# group1: data_size=2B, pad 4 DWORDs per 16 DWORDs (64B data -> 80B rows),
            // tensor_dim0=1152, tensor_dim1=256, tile_dim0=32, tile_dim1=128,
            // tensor_dim0_stride=1152 elements
            i32x8 g1;
            g1[0] = (1 << 16) | (1 << 20) | (3 << 22) | (3 << 25);
            g1[1] = (int)((unsigned)KTOT << 16);         // tensor_dim0[15:0] @ bits 63:48
            g1[2] = (int)((unsigned)COUT << 16);         // dim0[31:16]=0 | tensor_dim1[15:0]
            g1[3] = (int)(32u << 16);                    // tensor_dim1 hi = 0 | tile_dim0=32
            g1[4] = BM;                                  // tile_dim1=128 | tile_dim2=0
            g1[5] = KTOT;                                // tensor_dim0_stride[31:0]
            g1[6] = 0;                                   // stride hi | tensor_dim1_stride lo
            g1[7] = 0;
            i32x4 gz4 = {0, 0, 0, 0};
            i32x8 gz8 = {0, 0, 0, 0, 0, 0, 0, 0};
            __builtin_amdgcn_tensor_load_to_lds(g0, g1, gz4, gz4, gz8, 0);
        }
#else
        {
            const _Float16* srcA = wf + (size_t)(co_blk + st_row) * KTOT
                                      + (size_t)kc * BK + st_half * 16;
            v8h a0 = *(const v8h*)(srcA);
            v8h a1 = *(const v8h*)(srcA + 8);
            _Float16* dstA = ldsA + st_row * LDS_STRIDE + st_half * 16;
            *(v8h*)(dstA)     = a0;
            *(v8h*)(dstA + 8) = a1;
        }
#endif

        // ---- stage B: 128 spatial x 32 ci gathered from x, f32->f16 ----
        {
            const int iy = 2 * soy - 1 + ky;
            const int ix = 2 * sox - 1 + kx;
            const bool valid = ((unsigned)iy < HW) && ((unsigned)ix < HW);
            const float* src = x + (((size_t)simg * CIN + (ci0 + st_half * 16)) * HW + iy) * HW + ix;
            if (valid && (kc & 3) != 3)
                __builtin_prefetch(src + (size_t)32 * HW * HW, 0, 1);  // next ci block
            v8h h0, h1;
            #pragma unroll
            for (int j = 0; j < 8; ++j) {
                float v0 = valid ? src[(size_t)j       * HW * HW] : 0.0f;
                float v1 = valid ? src[(size_t)(j + 8) * HW * HW] : 0.0f;
                h0[j] = (_Float16)v0;
                h1[j] = (_Float16)v1;
            }
            _Float16* dst = ldsB + st_row * LDS_STRIDE + st_half * 16;
            *(v8h*)(dst)     = h0;
            *(v8h*)(dst + 8) = h1;
        }

#if USE_TDM
        if (wave == 0)
            __builtin_amdgcn_s_wait_tensorcnt(0);   // TDM done before publishing tile
#endif
        __syncthreads();

        // ---- fragment loads (16-bit A/B VGPR layout) + 8 WMMA ----
        // lanes 0-15 hold K {0..7}@+0, {16..23}@+16 ; lanes 16-31 hold
        // K {8..15}@+8, {24..31}@+24  (f16 element offsets)
        v16h afrag[2], bfrag[4];
        #pragma unroll
        for (int tm = 0; tm < 2; ++tm) {
            const _Float16* p = ldsA + (wave_m * 32 + tm * 16 + ln) * LDS_STRIDE + hi16 * 8;
            v8h lo = *(const v8h*)(p);
            v8h hi = *(const v8h*)(p + 16);
            afrag[tm] = __builtin_shufflevector(lo, hi,
                0,1,2,3,4,5,6,7,8,9,10,11,12,13,14,15);
        }
        #pragma unroll
        for (int tn = 0; tn < 4; ++tn) {
            const _Float16* p = ldsB + (wave_n * 64 + tn * 16 + ln) * LDS_STRIDE + hi16 * 8;
            v8h lo = *(const v8h*)(p);
            v8h hi = *(const v8h*)(p + 16);
            bfrag[tn] = __builtin_shufflevector(lo, hi,
                0,1,2,3,4,5,6,7,8,9,10,11,12,13,14,15);
        }
        #pragma unroll
        for (int tm = 0; tm < 2; ++tm)
            #pragma unroll
            for (int tn = 0; tn < 4; ++tn)
                acc[tm][tn] = __builtin_amdgcn_wmma_f32_16x16x32_f16(
                    false, afrag[tm], false, bfrag[tn],
                    (short)0, acc[tm][tn], false, false);
    }

    // ---- epilogue: bias + 2x2 nearest-neighbor replication, float2 stores ----
    #pragma unroll
    for (int tn = 0; tn < 4; ++tn) {
        const int s   = s_blk + wave_n * 64 + tn * 16 + ln;
        const int img = s / (OH * OW);
        int rem       = s - img * (OH * OW);
        const int oy  = rem / OW;
        const int ox  = rem - oy * OW;
        const size_t orow0 = (((size_t)img * COUT) * HW + 2 * oy) * HW + 2 * ox;
        #pragma unroll
        for (int tm = 0; tm < 2; ++tm) {
            #pragma unroll
            for (int r = 0; r < 8; ++r) {
                // C/D layout: vgpr r -> M = r + hi16*8, N = ln
                const int co = co_blk + wave_m * 32 + tm * 16 + r + hi16 * 8;
                const float v = acc[tm][tn][r] + bias[co];
                float2 p2 = make_float2(v, v);
                float* base = out + orow0 + (size_t)co * HW * HW;
                *(float2*)(base)      = p2;   // row 2*oy
                *(float2*)(base + HW) = p2;   // row 2*oy + 1
            }
        }
    }
}

extern "C" void kernel_launch(void* const* d_in, const int* in_sizes, int n_in,
                              void* d_out, int out_size, void* d_ws, size_t ws_size,
                              hipStream_t stream) {
    const float* x = (const float*)d_in[0];
    const float* w = (const float*)d_in[1];
    const float* b = (const float*)d_in[2];
    float* out     = (float*)d_out;
    _Float16* wf   = (_Float16*)d_ws;     // 256*1152*2 B = 576 KB scratch

    (void)in_sizes; (void)n_in; (void)out_size; (void)ws_size;

    repack_weight_f16<<<(COUT * KTOT + 255) / 256, 256, 0, stream>>>(w, wf);

    dim3 grid(NIMG * OH * OW / BN, COUT / BM);   // (784, 2)
    perf_conv_wmma<<<grid, 256, 0, stream>>>(x, wf, b, out);
}
